// LightweightGNN_18167711662741
// MI455X (gfx1250) — matrix operand order repaired and checked by reading, not compiled, not run on hardware
//
#include <hip/hip_runtime.h>
#include <hip/hip_bf16.h>

// ---------------------------------------------------------------------------
// LightweightGNN on MI455X (gfx1250, wave32, WMMA f32<=f16 16x16x32)
// N=1024, NODE_DIM=HID=128, EDGE_DIM=32, NREL=16
// ---------------------------------------------------------------------------

typedef __attribute__((ext_vector_type(16))) _Float16 v16h;
typedef __attribute__((ext_vector_type(8)))  _Float16 v8h;
typedef __attribute__((ext_vector_type(8)))  float    v8f;
typedef __attribute__((ext_vector_type(4)))  float    f32x4;

#define NN 1024
#define DD 128
#define EE 32
#define RR 16

__device__ __forceinline__ v8f wmma_f16(v16h a, v16h b, v8f c) {
  // D = A(16x32 f16) * B(32x16 f16) + C(16x16 f32)
  return __builtin_amdgcn_wmma_f32_16x16x32_f16(false, a, false, b,
                                                (short)0, c, false, false);
}

__device__ __forceinline__ v16h cat16(v8h a, v8h b) {
  return __builtin_shufflevector(a, b, 0,1,2,3,4,5,6,7,8,9,10,11,12,13,14,15);
}

// 8 consecutive f32 -> 8 f16
__device__ __forceinline__ v8h cvt8(const float* __restrict__ p) {
  f32x4 a = *(const f32x4*)p;
  f32x4 b = *(const f32x4*)(p + 4);
  v8h r;
  r[0]=(_Float16)a[0]; r[1]=(_Float16)a[1]; r[2]=(_Float16)a[2]; r[3]=(_Float16)a[3];
  r[4]=(_Float16)b[0]; r[5]=(_Float16)b[1]; r[6]=(_Float16)b[2]; r[7]=(_Float16)b[3];
  return r;
}
// B-operand: 16 consecutive f32 -> v16h
__device__ __forceinline__ v16h cvt16c(const float* p){ return cat16(cvt8(p), cvt8(p + 8)); }
// A-operand: two 8-float runs at +0 and +16 (K split {0..7,16..23}/{8..15,24..31})
__device__ __forceinline__ v16h cvt16r(const float* p){ return cat16(cvt8(p), cvt8(p + 16)); }
// f16 variants
__device__ __forceinline__ v16h ld16c(const _Float16* p){
  return cat16(*(const v8h*)p, *(const v8h*)(p + 8));
}
__device__ __forceinline__ v16h ld16r(const _Float16* p){
  return cat16(*(const v8h*)p, *(const v8h*)(p + 16));
}

// ---------------------------------------------------------------------------
// inv_deg[i] = 1 / (sum_j adj[i,j] + 1e-6)
__global__ __launch_bounds__(256) void k_deg(const float* __restrict__ adj,
                                             float* __restrict__ invdeg) {
  int i = blockIdx.x * 256 + threadIdx.x;
  if (i >= NN) return;
  const float* row = adj + (size_t)i * NN;
  float s = 0.f;
  for (int j = 0; j < NN; j += 4) {
    f32x4 v = *(const f32x4*)(row + j);
    s += v[0] + v[1] + v[2] + v[3];
  }
  invdeg[i] = 1.0f / (s + 1e-6f);
}

// node_feat -> f16 row-major + f16 transposed
__global__ __launch_bounds__(256) void k_trans(const float* __restrict__ src,
                                               _Float16* __restrict__ dstT,
                                               _Float16* __restrict__ dstRow) {
  int idx = blockIdx.x * 256 + threadIdx.x;   // 1024*128 total
  int i = idx >> 7, d = idx & 127;
  float v = src[idx];
  _Float16 h = (_Float16)v;
  dstRow[idx] = h;
  dstT[(size_t)d * NN + i] = h;
}

// ---------------------------------------------------------------------------
// aggT: C[d,i] = sum_j srcT[d,j] * adj[i,j]  -> dst[i,d] = C * invdeg[i] (f16)
__global__ __launch_bounds__(256) void k_aggT(const _Float16* __restrict__ srcT,
                                              const float* __restrict__ adj,
                                              const float* __restrict__ invdeg,
                                              _Float16* __restrict__ dst) {
  const int wv = threadIdx.x >> 5;
  const int lane = threadIdx.x & 31;
  const int ln = lane & 15, hi = lane >> 4;
  const int tile = blockIdx.x * 8 + wv;       // 512 tiles: 8 d-tiles x 64 i-tiles
  const int d0 = (tile & 7) * 16;
  const int i0 = (tile >> 3) * 16;
  const float scale = invdeg[i0 + ln];
  v8f acc = {};
  for (int kk = 0; kk < NN; kk += 32) {
    v16h a = ld16c(srcT + (size_t)(d0 + ln) * NN + kk + hi * 16);
    v16h b = cvt16c(adj + (size_t)(i0 + ln) * NN + kk + hi * 16);
    acc = wmma_f16(a, b, acc);
  }
#pragma unroll
  for (int r = 0; r < 8; r++) {
    int d = d0 + r + 8 * hi;
    dst[(size_t)(i0 + ln) * DD + d] = (_Float16)(acc[r] * scale);
  }
}

// ---------------------------------------------------------------------------
// layer: h_out = relu([pA | pB] @ W^T + b), W is [128 x 256] row-major.
// C[o,i]; store row-major f16 and transposed f16.
__global__ __launch_bounds__(256) void k_layer(const _Float16* __restrict__ pA,
                                               const _Float16* __restrict__ pB,
                                               const float* __restrict__ W,
                                               const float* __restrict__ bias,
                                               _Float16* __restrict__ outRow,
                                               _Float16* __restrict__ outT) {
  const int wv = threadIdx.x >> 5;
  const int lane = threadIdx.x & 31;
  const int ln = lane & 15, hi = lane >> 4;
  const int tile = blockIdx.x * 8 + wv;
  const int o0 = (tile & 7) * 16;
  const int i0 = (tile >> 3) * 16;
  v8f acc = {};
#pragma unroll
  for (int kb = 0; kb < 8; kb++) {
    const int kk = kb * 32;
    v16h a = cvt16c(W + (size_t)(o0 + ln) * 256 + kk + hi * 16);
    const _Float16* src = (kk < 128)
        ? (pA + (size_t)(i0 + ln) * DD + kk + hi * 16)
        : (pB + (size_t)(i0 + ln) * DD + (kk - 128) + hi * 16);
    v16h b = ld16c(src);
    acc = wmma_f16(a, b, acc);
  }
#pragma unroll
  for (int r = 0; r < 8; r++) {
    int o = o0 + r + 8 * hi;
    float v = acc[r] + bias[o];
    v = fmaxf(v, 0.0f);
    _Float16 hv = (_Float16)v;
    outRow[(size_t)(i0 + ln) * DD + o] = hv;
    outT[(size_t)o * NN + (i0 + ln)] = hv;
  }
}

// ---------------------------------------------------------------------------
// proj: Mat[i,h] = sum_d Wsub[h,d]*h2[i,d] (+ bias[h]); Wsub rows stride ldW.
__global__ __launch_bounds__(256) void k_proj(const float* __restrict__ Wsub, int ldW,
                                              const _Float16* __restrict__ hRow,
                                              const float* __restrict__ bias,
                                              float* __restrict__ Mat) {
  const int wv = threadIdx.x >> 5;
  const int lane = threadIdx.x & 31;
  const int ln = lane & 15, hi = lane >> 4;
  const int tile = blockIdx.x * 8 + wv;
  const int h0 = (tile & 7) * 16;
  const int i0 = (tile >> 3) * 16;
  v8f acc = {};
#pragma unroll
  for (int kb = 0; kb < 4; kb++) {
    const int kk = kb * 32;
    v16h a = cvt16c(Wsub + (size_t)(h0 + ln) * ldW + kk + hi * 16);
    v16h b = ld16c(hRow + (size_t)(i0 + ln) * DD + kk + hi * 16);
    acc = wmma_f16(a, b, acc);
  }
#pragma unroll
  for (int r = 0; r < 8; r++) {
    int h = h0 + r + 8 * hi;
    float v = acc[r] + (bias ? bias[h] : 0.0f);
    Mat[(size_t)(i0 + ln) * DD + h] = v;
  }
}

// ---------------------------------------------------------------------------
// edge kernel: per (i, 16-j tile) wave.
// GEMM1: pre[j,h] = edge[i,j,:]@We^T + Ai[h](incl b_c1) + Bj[j,h]; relu -> LDS(f16)
// GEMM2: logits[rel,j] = W_c2 @ pre^T, +b_c2, diag mask, store.
__global__ __launch_bounds__(256) void k_edge(const float* __restrict__ edge,
                                              const float* __restrict__ Amat,
                                              const float* __restrict__ Bmat,
                                              const float* __restrict__ Wc1,
                                              const float* __restrict__ Wc2,
                                              const float* __restrict__ bc2,
                                              float* __restrict__ out) {
  __shared__ _Float16 lds[8 * 16 * 32];       // per-wave 16j x 32h slab (8 KB)
  const int i  = blockIdx.x >> 3;
  const int jc = blockIdx.x & 7;
  const int wv = threadIdx.x >> 5;
  const int lane = threadIdx.x & 31;
  const int ln = lane & 15, hi = lane >> 4;
  const int j0 = jc * 128 + wv * 16;
  _Float16* ldsW = &lds[wv * 16 * 32];

  // loop-invariant: We^T B-tiles (8 x 16h), W_c2 A-tiles (K=128 in 4 chunks)
  v16h bwe[8];
#pragma unroll
  for (int t = 0; t < 8; t++)
    bwe[t] = cvt16c(Wc1 + (size_t)(t * 16 + ln) * 288 + 256 + hi * 16);
  v16h a2[4];
#pragma unroll
  for (int kb = 0; kb < 4; kb++)
    a2[kb] = cvt16c(Wc2 + (size_t)ln * 128 + kb * 32 + hi * 16);
  float bc2r[8], Ai[8];
#pragma unroll
  for (int r = 0; r < 8; r++) bc2r[r] = bc2[8 * hi + r];
#pragma unroll
  for (int t = 0; t < 8; t++) Ai[t] = Amat[(size_t)i * DD + t * 16 + ln];

  const int j = j0 + ln;                        // this lane's A-row & output col
  v16h aedge = cvt16c(edge + ((size_t)i * NN + j) * EE + hi * 16);

  v8f c2 = {};
#pragma unroll
  for (int tt = 0; tt < 4; tt++) {
#pragma unroll
    for (int s = 0; s < 2; s++) {
      const int t = tt * 2 + s;
      v8f c = {};
      c = wmma_f16(aedge, bwe[t], c);
#pragma unroll
      for (int r = 0; r < 8; r++) {
        int jr = j0 + r + 8 * hi;
        float bj = Bmat[(size_t)jr * DD + t * 16 + ln];
        float pre = c[r] + Ai[t] + bj;
        pre = fmaxf(pre, 0.0f);
        ldsW[(r + 8 * hi) * 32 + s * 16 + ln] = (_Float16)pre;
      }
    }
    asm volatile("s_wait_dscnt 0" ::: "memory"); // same-wave LDS RAW
    const v8h* lp = (const v8h*)&ldsW[ln * 32 + hi * 16];
    v16h b2 = cat16(lp[0], lp[1]);
    c2 = wmma_f16(a2[tt], b2, c2);
    asm volatile("" ::: "memory");               // keep next stores after loads
  }

  const float m = (i == j) ? 0.0f : 1.0f;
  f32x4 o1, o2;
  o1[0] = m * (c2[0] + bc2r[0]); o1[1] = m * (c2[1] + bc2r[1]);
  o1[2] = m * (c2[2] + bc2r[2]); o1[3] = m * (c2[3] + bc2r[3]);
  o2[0] = m * (c2[4] + bc2r[4]); o2[1] = m * (c2[5] + bc2r[5]);
  o2[2] = m * (c2[6] + bc2r[6]); o2[3] = m * (c2[7] + bc2r[7]);
  float* op = out + ((size_t)i * NN + j) * RR + 8 * hi;
  *(f32x4*)op = o1;
  *(f32x4*)(op + 4) = o2;
}

// ---------------------------------------------------------------------------
extern "C" void kernel_launch(void* const* d_in, const int* in_sizes, int n_in,
                              void* d_out, int out_size, void* d_ws, size_t ws_size,
                              hipStream_t stream) {
  const float* node_feat = (const float*)d_in[0];
  const float* edge_feat = (const float*)d_in[1];
  const float* adj       = (const float*)d_in[2];
  const float* W_g1      = (const float*)d_in[3];
  const float* b_g1      = (const float*)d_in[4];
  const float* W_g2      = (const float*)d_in[5];
  const float* b_g2      = (const float*)d_in[6];
  const float* W_c1      = (const float*)d_in[7];
  const float* b_c1      = (const float*)d_in[8];
  const float* W_c2      = (const float*)d_in[9];
  const float* b_c2      = (const float*)d_in[10];
  float* out = (float*)d_out;

  char* wp = (char*)d_ws;
  auto take = [&](size_t bytes) -> char* {
    char* p = wp;
    wp += (bytes + 255) & ~(size_t)255;
    return p;
  };
  float*     invdeg  = (float*)take(NN * sizeof(float));
  _Float16*  featT   = (_Float16*)take((size_t)NN * DD * 2);
  _Float16*  featRow = (_Float16*)take((size_t)NN * DD * 2);
  _Float16*  agg1    = (_Float16*)take((size_t)NN * DD * 2);
  _Float16*  h1Row   = (_Float16*)take((size_t)NN * DD * 2);
  _Float16*  h1T     = (_Float16*)take((size_t)NN * DD * 2);
  _Float16*  agg2    = (_Float16*)take((size_t)NN * DD * 2);
  _Float16*  h2Row   = (_Float16*)take((size_t)NN * DD * 2);
  _Float16*  h2T     = (_Float16*)take((size_t)NN * DD * 2);  // unused sink
  float*     Amat    = (float*)take((size_t)NN * DD * 4);
  float*     Bmat    = (float*)take((size_t)NN * DD * 4);

  k_deg  <<<4,    256, 0, stream>>>(adj, invdeg);
  k_trans<<<512,  256, 0, stream>>>(node_feat, featT, featRow);
  k_aggT <<<64,   256, 0, stream>>>(featT, adj, invdeg, agg1);
  k_layer<<<64,   256, 0, stream>>>(featRow, agg1, W_g1, b_g1, h1Row, h1T);
  k_aggT <<<64,   256, 0, stream>>>(h1T, adj, invdeg, agg2);
  k_layer<<<64,   256, 0, stream>>>(h1Row, agg2, W_g2, b_g2, h2Row, h2T);
  k_proj <<<64,   256, 0, stream>>>(W_c1,       2 * DD + EE, h2Row, b_c1,    Amat);
  k_proj <<<64,   256, 0, stream>>>(W_c1 + DD,  2 * DD + EE, h2Row, nullptr, Bmat);
  k_edge <<<8192, 256, 0, stream>>>(edge_feat, Amat, Bmat, W_c1, W_c2, b_c2, out);
}
